// Attention_77876347011400
// MI455X (gfx1250) — matrix operand order repaired and checked by reading, not compiled, not run on hardware
//
#include <hip/hip_runtime.h>
#include <hip/hip_bf16.h>

typedef __attribute__((ext_vector_type(16))) _Float16 v16h;
typedef __attribute__((ext_vector_type(8)))  _Float16 v8h;
typedef __attribute__((ext_vector_type(8)))  float    v8f;

union U16h { v16h v; v8h h[2]; };

constexpr int kB = 2, kN = 2048, kDIM = 1024, kH = 16, kDH = 64, kDI = 1024;
constexpr int kBN = kB * kN;                 // 4096 total rows
constexpr float kScale = 0.125f;             // DH^-0.5
constexpr float kNegBig = -1e30f;

// ---- DPP16 16-lane max reduction (VALU only; no LDS, no counters) ----------
template <int CTRL>
__device__ __forceinline__ float dpp_mov_f32(float x) {
  return __int_as_float(
      __builtin_amdgcn_update_dpp(0, __float_as_int(x), CTRL, 0xf, 0xf, true));
}
__device__ __forceinline__ float rowmax16(float m) {
  m = fmaxf(m, dpp_mov_f32<0x0B1>(m));  // quad_perm(1,0,3,2) : xor 1
  m = fmaxf(m, dpp_mov_f32<0x04E>(m));  // quad_perm(2,3,0,1) : xor 2
  m = fmaxf(m, dpp_mov_f32<0x140>(m));  // row_mirror  : quad q <-> 3-q
  m = fmaxf(m, dpp_mov_f32<0x141>(m));  // row_half_mirror : combine pairs
  return m;                             // every lane of the 16-lane row = max
}

// ---------------------------------------------------------------- converts
__global__ void cvt_f32_to_f16(const float* __restrict__ in,
                               _Float16* __restrict__ out, int n) {
  int i = blockIdx.x * blockDim.x + threadIdx.x;
  if (i < n) out[i] = (_Float16)in[i];
}

// out[n*K + k] = (f16) in[k*Nout + n]   (in: [K x Nout] f32 row-major)
__global__ void transpose_cvt(const float* __restrict__ in,
                              _Float16* __restrict__ out, int K, int Nout) {
  int i = blockIdx.x * blockDim.x + threadIdx.x;
  if (i >= K * Nout) return;
  int k = i / Nout, n = i - k * Nout;
  out[(size_t)n * K + k] = (_Float16)in[i];
}

// Vt[((b*H+h)*DH + d)*N + n] = KVh[(b*N+n)*2*DI + DI + h*DH + d]
__global__ void repack_v(const _Float16* __restrict__ kvh,
                         _Float16* __restrict__ vt) {
  int i = blockIdx.x * blockDim.x + threadIdx.x;   // over B*H*DH*N, n fastest
  if (i >= kB * kH * kDH * kN) return;
  int n = i & (kN - 1);
  int d = (i >> 11) & (kDH - 1);
  int h = (i >> 17) & (kH - 1);
  int b = i >> 21;
  vt[i] = kvh[(size_t)(b * kN + n) * (2 * kDI) + kDI + h * kDH + d];
}

// ---------------------------------------------------------------- GEMM
// C[M x Nout] = A[M x K] * B, B pre-transposed as Bt[Nout x K] so every WMMA
// operand load is a contiguous 16B-aligned per-lane vector load.
// One wave owns a 32x64 C tile: 2 A-fragments x 4 B-fragments = 8 WMMAs/k-step.
template <typename OutT>
__global__ void gemm_wmma_f16(const _Float16* __restrict__ A,
                              const _Float16* __restrict__ Bt,
                              OutT* __restrict__ C,
                              int M, int K, int Nout) {
  const int wid  = (blockIdx.x * blockDim.x + threadIdx.x) >> 5;
  const int lane = threadIdx.x & 31;
  const int Mt = M >> 5, Nt = Nout >> 6;
  if (wid >= Mt * Nt) return;
  const int m0 = (wid % Mt) << 5;   // consecutive waves share the B tile
  const int n0 = (wid / Mt) << 6;
  const int lm = lane & 15;
  const int kh = lane >> 4;

  v8f acc0[4] = {}, acc1[4] = {};
  const _Float16* arow0 = A + (size_t)(m0 + lm) * K;
  const _Float16* arow1 = A + (size_t)(m0 + 16 + lm) * K;

  for (int k0 = 0; k0 < K; k0 += 32) {
    U16h a0, a1;                              // A 16x32: K = j + 8*((j>=8)+kh)
    a0.h[0] = *(const v8h*)(arow0 + k0 + 8 * kh);
    a0.h[1] = *(const v8h*)(arow0 + k0 + 16 + 8 * kh);
    a1.h[0] = *(const v8h*)(arow1 + k0 + 8 * kh);
    a1.h[1] = *(const v8h*)(arow1 + k0 + 16 + 8 * kh);
    __builtin_prefetch(arow0 + k0 + 128, 0, 1);   // -> global_prefetch_b8
    __builtin_prefetch(arow1 + k0 + 128, 0, 1);
#pragma unroll
    for (int t = 0; t < 4; ++t) {             // B 32x16: K = j + 8*((j>=8)+kh), N = lm
      const _Float16* bp = Bt + (size_t)(n0 + 16 * t + lm) * K + k0 + 8 * kh;
      U16h bf;
      bf.h[0] = *(const v8h*)(bp);
      bf.h[1] = *(const v8h*)(bp + 16);
      __builtin_prefetch(bp + 128, 0, 1);
      acc0[t] = __builtin_amdgcn_wmma_f32_16x16x32_f16(
          false, a0.v, false, bf.v, (short)0, acc0[t], false, false);
      acc1[t] = __builtin_amdgcn_wmma_f32_16x16x32_f16(
          false, a1.v, false, bf.v, (short)0, acc1[t], false, false);
    }
  }
#pragma unroll
  for (int t = 0; t < 4; ++t) {
    v8f av0 = acc0[t], av1 = acc1[t];
#pragma unroll
    for (int r = 0; r < 8; ++r) {             // C: M = r + 8*kh, N = n0+16t+lm
      C[(size_t)(m0 + r + 8 * kh) * Nout + n0 + 16 * t + lm]      = (OutT)av0[r];
      C[(size_t)(m0 + 16 + r + 8 * kh) * Nout + n0 + 16 * t + lm] = (OutT)av1[r];
    }
  }
}

// ---------------------------------------------------------------- attention
// One wave per (b, h, 16-query tile); flash-style online softmax, 32 keys/iter.
// Mask folded into the WMMA C operand; row-sum via WMMA against ones;
// row-max via DPP16 butterfly (no LDS ops in the reduction).
__global__ void attn_fused(const _Float16* __restrict__ Qh,
                           const _Float16* __restrict__ KVh,
                           const _Float16* __restrict__ Vt,
                           const unsigned char* __restrict__ mask, // jnp bool = 1B
                           _Float16* __restrict__ Oh) {
  const int qt = blockIdx.x & ((kN >> 4) - 1);       // 128 q-tiles
  const int h  = (blockIdx.x >> 7) & (kH - 1);
  const int b  = blockIdx.x >> 11;
  const int lane = threadIdx.x & 31;
  const int lm = lane & 15, kh = lane >> 4;
  const int q0 = qt << 4;

  __shared__ _Float16 ldsP[16 * 32] __attribute__((aligned(16)));

  // Q as two A-fragments over DH=64, with softmax scale pre-folded (once).
  const _Float16* qrow = Qh + (size_t)(b * kN + q0 + lm) * kDI + h * kDH;
  U16h qa0, qa1;
  qa0.h[0] = *(const v8h*)(qrow + 16 * kh);
  qa0.h[1] = *(const v8h*)(qrow + 16 * kh + 8);
  qa1.h[0] = *(const v8h*)(qrow + 32 + 16 * kh);
  qa1.h[1] = *(const v8h*)(qrow + 32 + 16 * kh + 8);
#pragma unroll
  for (int j = 0; j < 16; ++j) {
    qa0.v[j] = qa0.v[j] * (_Float16)kScale;
    qa1.v[j] = qa1.v[j] * (_Float16)kScale;
  }

  U16h ones;                                  // B-matrix of ones for row-sums
#pragma unroll
  for (int j = 0; j < 16; ++j) ones.v[j] = (_Float16)1.0f;

  const _Float16* kbase = KVh + (size_t)(b * kN) * (2 * kDI) + h * kDH;
  const _Float16* vbase = Vt + (size_t)((b * kH + h) * kDH) * kN;
  const unsigned char* mp = mask + b * kN;

  float mx[8], ls[8];
#pragma unroll
  for (int r = 0; r < 8; ++r) { mx[r] = kNegBig; ls[r] = 0.f; }
  v8f oacc[4] = {};

  for (int n0 = 0; n0 < kN; n0 += 32) {
    // mask -> additive bias, splat into the S accumulator (key idx = lane only)
    const float bias0 = mp[n0 + lm]      ? 0.f : kNegBig;
    const float bias1 = mp[n0 + 16 + lm] ? 0.f : kNegBig;
    v8f s0, s1;
#pragma unroll
    for (int r = 0; r < 8; ++r) { s0[r] = bias0; s1[r] = bias1; }

    // S = (Q*scale) K^T + bias : two 16-key tiles, contiguous K rows per lane.
    const _Float16* k0p = kbase + (size_t)(n0 + lm) * (2 * kDI);
    const _Float16* k1p = kbase + (size_t)(n0 + 16 + lm) * (2 * kDI);
    __builtin_prefetch(k0p + 32 * 2 * kDI, 0, 1);
    __builtin_prefetch(k1p + 32 * 2 * kDI, 0, 1);
    U16h kb;
    kb.h[0] = *(const v8h*)(k0p + 16 * kh);
    kb.h[1] = *(const v8h*)(k0p + 16 * kh + 8);
    s0 = __builtin_amdgcn_wmma_f32_16x16x32_f16(false, qa0.v, false, kb.v, (short)0, s0, false, false);
    kb.h[0] = *(const v8h*)(k0p + 32 + 16 * kh);
    kb.h[1] = *(const v8h*)(k0p + 32 + 16 * kh + 8);
    s0 = __builtin_amdgcn_wmma_f32_16x16x32_f16(false, qa1.v, false, kb.v, (short)0, s0, false, false);
    kb.h[0] = *(const v8h*)(k1p + 16 * kh);
    kb.h[1] = *(const v8h*)(k1p + 16 * kh + 8);
    s1 = __builtin_amdgcn_wmma_f32_16x16x32_f16(false, qa0.v, false, kb.v, (short)0, s1, false, false);
    kb.h[0] = *(const v8h*)(k1p + 32 + 16 * kh);
    kb.h[1] = *(const v8h*)(k1p + 32 + 16 * kh + 8);
    s1 = __builtin_amdgcn_wmma_f32_16x16x32_f16(false, qa1.v, false, kb.v, (short)0, s1, false, false);

    // online-softmax: row max via DPP16 butterfly; exp; P to LDS (f16)
    float corr[8];
#pragma unroll
    for (int r = 0; r < 8; ++r) {            // row = r + 8*kh, in 16-lane group
      float m = rowmax16(fmaxf(s0[r], s1[r]));
      float nm = fmaxf(mx[r], m);
      corr[r] = __expf(mx[r] - nm);
      mx[r] = nm;
      // masked logits sit at -1e30 -> exp underflows to exactly 0
      ldsP[(r + 8 * kh) * 32 + lm]      = (_Float16)__expf(s0[r] - nm);
      ldsP[(r + 8 * kh) * 32 + 16 + lm] = (_Float16)__expf(s1[r] - nm);
    }
    __syncthreads();
    U16h pa;                                 // P as A-fragment (C->A layout swap)
    pa.h[0] = *(const v8h*)(&ldsP[lm * 32 + 16 * kh]);
    pa.h[1] = *(const v8h*)(&ldsP[lm * 32 + 16 * kh + 8]);
    __syncthreads();

    // row sums on the matrix core: rs = P x ones (every column = row sum)
    v8f z = {};
    v8f rs = __builtin_amdgcn_wmma_f32_16x16x32_f16(
        false, pa.v, false, ones.v, (short)0, z, false, false);
#pragma unroll
    for (int r = 0; r < 8; ++r) ls[r] = ls[r] * corr[r] + rs[r];

    // O = diag(corr) * O + P * V  (4 d-tiles; Vt rows contiguous per lane)
#pragma unroll
    for (int t = 0; t < 4; ++t) {
      v8f c;
#pragma unroll
      for (int r = 0; r < 8; ++r) c[r] = oacc[t][r] * corr[r];
      const _Float16* vp = vbase + (size_t)(16 * t + lm) * kN + n0 + 16 * kh;
      U16h vb;
      vb.h[0] = *(const v8h*)(vp);
      vb.h[1] = *(const v8h*)(vp + 8);
      oacc[t] = __builtin_amdgcn_wmma_f32_16x16x32_f16(
          false, pa.v, false, vb.v, (short)0, c, false, false);
    }
  }

  // normalize (reciprocal once per row) and store into [b, n, h*d]
  float inv[8];
#pragma unroll
  for (int r = 0; r < 8; ++r) inv[r] = 1.0f / ls[r];
#pragma unroll
  for (int t = 0; t < 4; ++t) {
    v8f av = oacc[t];
#pragma unroll
    for (int r = 0; r < 8; ++r) {
      Oh[(size_t)(b * kN + q0 + r + 8 * kh) * kDI + h * kDH + 16 * t + lm] =
          (_Float16)(av[r] * inv[r]);
    }
  }
}

// ---------------------------------------------------------------- launch
extern "C" void kernel_launch(void* const* d_in, const int* in_sizes, int n_in,
                              void* d_out, int out_size, void* d_ws, size_t ws_size,
                              hipStream_t stream) {
  const float* x   = (const float*)d_in[0];
  const unsigned char* mask = (const unsigned char*)d_in[1];  // bool array
  const float* Wq  = (const float*)d_in[2];
  const float* Wkv = (const float*)d_in[3];
  const float* Wo  = (const float*)d_in[4];
  float* out = (float*)d_out;

  // workspace carve-up (f16; ~56 MB total)
  _Float16* p = (_Float16*)d_ws;
  _Float16* xh   = p;  p += (size_t)kBN * kDIM;
  _Float16* WqT  = p;  p += (size_t)kDIM * kDI;        // [DI  x DIM]
  _Float16* WkvT = p;  p += (size_t)kDIM * 2 * kDI;    // [2DI x DIM]
  _Float16* WoT  = p;  p += (size_t)kDI * kDIM;        // [DIM x DI]
  _Float16* Qh   = p;  p += (size_t)kBN * kDI;
  _Float16* KVh  = p;  p += (size_t)kBN * 2 * kDI;
  _Float16* Vt   = p;  p += (size_t)kB * kH * kDH * kN;
  _Float16* Ohb  = p;

  {
    int n = kBN * kDIM;
    cvt_f32_to_f16<<<(n + 255) / 256, 256, 0, stream>>>(x, xh, n);
  }
  transpose_cvt<<<(kDIM * kDI + 255) / 256, 256, 0, stream>>>(Wq, WqT, kDIM, kDI);
  transpose_cvt<<<(kDIM * 2 * kDI + 255) / 256, 256, 0, stream>>>(Wkv, WkvT, kDIM, 2 * kDI);
  transpose_cvt<<<(kDI * kDIM + 255) / 256, 256, 0, stream>>>(Wo, WoT, kDI, kDIM);

  {  // Q = x @ Wq  (f16 out)
    int waves = (kBN / 32) * (kDI / 64);
    gemm_wmma_f16<_Float16><<<(waves * 32 + 255) / 256, 256, 0, stream>>>(
        xh, WqT, Qh, kBN, kDIM, kDI);
  }
  {  // KV = x @ Wkv
    int waves = (kBN / 32) * (2 * kDI / 64);
    gemm_wmma_f16<_Float16><<<(waves * 32 + 255) / 256, 256, 0, stream>>>(
        xh, WkvT, KVh, kBN, kDIM, 2 * kDI);
  }
  {
    int n = kB * kH * kDH * kN;
    repack_v<<<(n + 255) / 256, 256, 0, stream>>>(KVh, Vt);
  }

  attn_fused<<<kB * kH * (kN / 16), 32, 0, stream>>>(Qh, KVh, Vt, mask, Ohb);

  {  // out = O @ Wo  (f32 out)
    int waves = (kBN / 32) * (kDIM / 64);
    gemm_wmma_f16<float><<<(waves * 32 + 255) / 256, 256, 0, stream>>>(
        Ohb, WoT, out, kBN, kDI, kDIM);
  }
}
